// SingleScaleDepthFusionSpatialEnhancer_69148973466262
// MI455X (gfx1250) — matrix operand order repaired and checked by reading, not compiled, not run on hardware
//
#include <hip/hip_runtime.h>

typedef __attribute__((ext_vector_type(16))) __bf16 v16bf;
typedef __attribute__((ext_vector_type(8)))  float  v8f;
typedef unsigned int u32x4 __attribute__((ext_vector_type(4)));
typedef int          i32x8 __attribute__((ext_vector_type(8)));
typedef int          i32x4 __attribute__((ext_vector_type(4)));

#define B_   4
#define C_   256
#define N_   4096
#define M_   16384
#define D_   256
#define F3_  32
#define FUS_ 288
#define FF_  1024

#if __has_builtin(__builtin_amdgcn_tensor_load_to_lds) && __has_builtin(__builtin_amdgcn_s_wait_tensorcnt)
#define HAVE_TDM 1
#else
#define HAVE_TDM 0
#endif

#define STR2(x) #x
#define STR(x) STR2(x)
#if HAVE_TDM
#pragma message("CDNA5 probe: TDM path ENABLED, clang major = " STR(__clang_major__))
#else
#pragma message("CDNA5 probe: TDM path DISABLED (builtin missing), clang major = " STR(__clang_major__))
#endif

static __device__ __forceinline__ unsigned short f2bf(float f) {
    unsigned u = __float_as_uint(f);
    unsigned r = 0x7FFFu + ((u >> 16) & 1u);   // round-to-nearest-even
    u += r;
    return (unsigned short)(u >> 16);
}
static __device__ __forceinline__ float bf2f(unsigned short h) {
    return __uint_as_float(((unsigned)h) << 16);
}

// ---------------- weight f32 -> bf16 ----------------
__global__ void k_f32_to_bf16(const float* __restrict__ s, unsigned short* __restrict__ d, int n) {
    int i = blockIdx.x * 256 + threadIdx.x;
    if (i < n) d[i] = f2bf(s[i]);
}

// ---------------- feat (B,C,N) -> Xf[(b*N+n)*288 + c] bf16 ----------------
__global__ void k_feat_transpose(const float* __restrict__ feat, unsigned short* __restrict__ Xf) {
    __shared__ float tile[32][33];
    int n0 = blockIdx.x * 32, c0 = blockIdx.y * 32, b = blockIdx.z;
    int tx = threadIdx.x, ty = threadIdx.y;
#pragma unroll
    for (int i = 0; i < 4; i++) {
        int c = c0 + ty + i * 8;
        tile[ty + i * 8][tx] = feat[((size_t)b * C_ + c) * N_ + n0 + tx];
    }
    __syncthreads();
#pragma unroll
    for (int i = 0; i < 4; i++) {
        int n = n0 + ty + i * 8;
        int c = c0 + tx;
        Xf[((size_t)(b * N_ + n)) * FUS_ + c] = f2bf(tile[tx][ty + i * 8]);
    }
}

// ---------------- depth reduction + pts feature: one wave32 per token ----------------
__global__ void k_depth_pts(const float* __restrict__ dp, const float* __restrict__ Kmat,
                            const float* __restrict__ pw, const float* __restrict__ pb,
                            const int* __restrict__ stride_p, unsigned short* __restrict__ Xf) {
    int wid = threadIdx.x >> 5, lane = threadIdx.x & 31;
    int t = blockIdx.x * 8 + wid;
    const float* dprow = dp + (size_t)t * D_;
    float s0 = 0.f, s1 = 0.f;
#pragma unroll
    for (int i = 0; i < 8; i++) {
        int d = lane + 32 * i;
        float v = dprow[d];
        if (!(fabsf(v) < 3.0e38f)) v = 0.f;     // nan/inf -> 0
        v = fmaxf(v, 1e-6f);
        s0 += v;
        s1 += v * (0.2f + (float)d * (0.8f / 255.f));
    }
#pragma unroll
    for (int off = 16; off; off >>= 1) {
        s0 += __shfl_xor(s0, off, 32);
        s1 += __shfl_xor(s1, off, 32);
    }
    float zbar = s1 / s0;                        // normalized probs sum to 1
    int b = t >> 12, n = t & (N_ - 1);
    int w = n & 63, h = n >> 6;
    float S  = (float)stride_p[0];
    float u  = ((float)w + 0.5f) * S - 0.5f;
    float vv = ((float)h + 0.5f) * S - 0.5f;
    float fx = Kmat[b * 9 + 0], fy = Kmat[b * 9 + 4];
    float cx = Kmat[b * 9 + 2], cy = Kmat[b * 9 + 5];
    float ax = (u - cx) / fx, ay = (vv - cy) / fy;
    int k = lane;
    float sk  = ax * pw[k * 3 + 0] + ay * pw[k * 3 + 1] + pw[k * 3 + 2];
    float val = zbar * sk + pb[k];
    Xf[(size_t)t * FUS_ + C_ + k] = f2bf(val);
}

#if HAVE_TDM
// TDM: DMA a 128-row x 32-col bf16 tile (row stride = K elems) into LDS at lds_off.
// D# per CDNA5 ISA 8.3/8.4: group0 = {flags, lds_addr, global_addr, type=2},
// group1 = {data_size=2B, tensor dims (huge, never OOB), tile 32x128, dim0 stride=K}.
static __device__ __forceinline__ void tdm_load_a(unsigned lds_off,
                                                  const unsigned short* gsrc, int K) {
    unsigned long long ga = (unsigned long long)(const void*)gsrc;
    u32x4 g0;
    g0[0] = 1u;                                         // count=1, user descriptor
    g0[1] = lds_off;                                    // LDS byte address
    g0[2] = (unsigned)ga;                               // global_addr[31:0]
    g0[3] = (unsigned)((ga >> 32) & 0x01FFFFFFu) | (2u << 30);  // addr[56:32], type=2
    const unsigned td0 = 1u << 20, td1 = 1u << 20;      // tensor dims (no clamp)
    i32x8 g1;
    g1[0] = (int)(1u << 16);                            // data_size=1 (2 bytes), wg_mask=0
    g1[1] = (int)((td0 & 0xFFFFu) << 16);               // tensor_dim0[15:0]
    g1[2] = (int)(((td0 >> 16) & 0xFFFFu) | ((td1 & 0xFFFFu) << 16));
    g1[3] = (int)(((td1 >> 16) & 0xFFFFu) | (32u << 16)); // tile_dim0 = 32 elems
    g1[4] = 128;                                        // tile_dim1 = 128 rows, tile_dim2=0
    g1[5] = K;                                          // tensor_dim0_stride[31:0]
    g1[6] = 0;                                          // stride hi, dim1_stride lo
    g1[7] = 0;
    i32x4 gz = {0, 0, 0, 0};
#if defined(__clang_major__) && __clang_major__ >= 23
    i32x8 gz8 = {0, 0, 0, 0, 0, 0, 0, 0};
    __builtin_amdgcn_tensor_load_to_lds(g0, g1, gz, gz, gz8, 0);
#else
    __builtin_amdgcn_tensor_load_to_lds(g0, g1, gz, gz, 0);
#endif
}
#endif

// ---------------- bf16 WMMA GEMM: Y[M,Nn] = act(X[M,K] @ Wt[Nn,K]^T + bias) ----------------
// block = 8 waves: 128(M) x 128(N); wave = 32 x 64 (2 row-groups x 4 col-tiles = 8 accums).
// K step 32. A tiles: TDM double-buffered LDS. B fragments: 2 explicit register sets,
// K-loop manually unrolled x2 -> no rotation copies. B reused by both row-groups.
// mode 0: relu -> bf16 ; mode 1: bias -> bf16 ; mode 2: bias + resid(bf16) -> f32
__global__ void __launch_bounds__(256)
k_gemm_bf16(const unsigned short* __restrict__ X, const unsigned short* __restrict__ Wt,
            const float* __restrict__ bias, unsigned short* __restrict__ Ybf,
            float* __restrict__ Yf32, const unsigned short* __restrict__ resid,
            int Nn, int K, int mode) {
    __shared__ __align__(16) unsigned short ldsA[2][128 * 32];   // 2 x 8KB
    const int m0    = blockIdx.y * 128;
    const int nBase = blockIdx.x * 128;
    const int wid   = threadIdx.x >> 5, lane = threadIdx.x & 31;
    const int waveM = wid & 3, waveN = wid >> 2;
    const int khalf = lane >> 4, l15 = lane & 15;
    const int KT    = K >> 5;

    typedef union { uint4 u[2]; v16bf v; } FragU;

    v8f acc[2][4];
    v8f vz = {0.f, 0.f, 0.f, 0.f, 0.f, 0.f, 0.f, 0.f};
    int  colB[4];
    bool valid[4];
    const unsigned short* bptr[4];
#pragma unroll
    for (int a = 0; a < 4; a++) {
        acc[0][a] = vz;
        acc[1][a] = vz;
        colB[a]  = nBase + waveN * 64 + a * 16;
        valid[a] = (colB[a] < Nn);                       // wave-uniform (Nn % 16 == 0)
        int cc   = valid[a] ? colB[a] : (Nn - 16);       // clamp: branch-free in-bounds loads
        bptr[a]  = Wt + (size_t)(cc + l15) * K + 8 * khalf;
    }

#if HAVE_TDM
    const unsigned ldsOff0 = (unsigned)(unsigned long long)(const void*)&ldsA[0][0];
    const unsigned ldsOff1 = ldsOff0 + 128 * 32 * 2;
#endif

    // stage(kt): make LDS tile for step kt ready; kick DMA for kt+1
    auto stage = [&](int kt) -> const unsigned short* {
#if HAVE_TDM
        if (wid == 0) __builtin_amdgcn_s_wait_tensorcnt(0);      // tile kt landed
        __syncthreads();                                         // publish; all done with buf kt-1
        if (wid == 0 && kt + 1 < KT)                             // DMA kt+1 under compute of kt
            tdm_load_a(((kt + 1) & 1) ? ldsOff1 : ldsOff0,
                       X + (size_t)m0 * K + (kt + 1) * 32, K);
        return &ldsA[kt & 1][0];
#else
        __syncthreads();
#pragma unroll
        for (int rep = 0; rep < 2; rep++) {                      // 512 x 16B chunks / 256 thr
            int idx = threadIdx.x + 256 * rep;
            int row = idx >> 2, ch = idx & 3;
            *(uint4*)&ldsA[0][row * 32 + ch * 8] =
                *(const uint4*)(X + (size_t)(m0 + row) * K + kt * 32 + ch * 8);
        }
        __syncthreads();
        return &ldsA[0][0];
#endif
    };
    auto loadA = [&](const unsigned short* abase, FragU& a0, FragU& a1) {
        const unsigned short* ar0 = abase + (waveM * 32 + l15) * 32;
        const unsigned short* ar1 = ar0 + 16 * 32;
        a0.u[0] = *(const uint4*)(ar0 + 8 * khalf);
        a0.u[1] = *(const uint4*)(ar0 + 16 + 8 * khalf);
        a1.u[0] = *(const uint4*)(ar1 + 8 * khalf);
        a1.u[1] = *(const uint4*)(ar1 + 16 + 8 * khalf);
    };
    auto loadB = [&](FragU* bb, int kt) {
#pragma unroll
        for (int a = 0; a < 4; a++) {
            const unsigned short* p = bptr[a] + kt * 32;
            bb[a].u[0] = *(const uint4*)(p);
            bb[a].u[1] = *(const uint4*)(p + 16);
        }
    };
    auto mma = [&](FragU& a0, FragU& a1, FragU* bb) {
#pragma unroll
        for (int a = 0; a < 4; a++) {
            acc[0][a] = __builtin_amdgcn_wmma_f32_16x16x32_bf16(
                false, a0.v, false, bb[a].v, (short)0, acc[0][a], false, false);
            acc[1][a] = __builtin_amdgcn_wmma_f32_16x16x32_bf16(
                false, a1.v, false, bb[a].v, (short)0, acc[1][a], false, false);
        }
    };

    FragU b0[4], b1[4], a0, a1;
    loadB(b0, 0);                                       // prime B pipeline
#if HAVE_TDM
    if (wid == 0) tdm_load_a(ldsOff0, X + (size_t)m0 * K, K);    // prologue: tile 0
#endif

    for (int kt = 0; kt < KT;) {
        const unsigned short* ab = stage(kt);
        loadA(ab, a0, a1);
        if (kt + 1 < KT) loadB(b1, kt + 1);
        mma(a0, a1, b0);
        ++kt;
        if (kt >= KT) break;
        ab = stage(kt);
        loadA(ab, a0, a1);
        if (kt + 1 < KT) loadB(b0, kt + 1);
        mma(a0, a1, b1);
        ++kt;
    }

    // epilogue: C/D layout -> row = r + 8*khalf, col = l15
#pragma unroll
    for (int g = 0; g < 2; g++) {
        const int rowBase = m0 + waveM * 32 + g * 16 + (khalf << 3);
#pragma unroll
        for (int a = 0; a < 4; a++) {
            if (!valid[a]) continue;
            int col = colB[a] + l15;
            float bv = bias[col];
#pragma unroll
            for (int r = 0; r < 8; r++) {
                int row = rowBase + r;
                float v = acc[g][a][r] + bv;
                if (mode == 0) v = fmaxf(v, 0.f);
                if (mode == 2) {
                    v += bf2f(resid[(size_t)row * FUS_ + col]);
                    Yf32[(size_t)row * Nn + col] = v;
                } else {
                    Ybf[(size_t)row * Nn + col] = f2bf(v);
                }
            }
        }
    }
}

// ---------------- LayerNorm over C=256 + transpose to (B,C,N) ----------------
__global__ void __launch_bounds__(256)
k_ln_transpose(const float* __restrict__ Y, const float* __restrict__ g,
               const float* __restrict__ bb, float* __restrict__ outp) {
    __shared__ float tile[32][C_ + 1];
    int t0 = blockIdx.x * 32;
#pragma unroll 4
    for (int i = 0; i < 32; i++) {
        int idx = threadIdx.x + 256 * i;
        int row = idx >> 8, col = idx & 255;
        tile[row][col] = Y[(size_t)(t0 + row) * C_ + col];
    }
    __syncthreads();
    int wid = threadIdx.x >> 5, lane = threadIdx.x & 31;
#pragma unroll
    for (int rr = 0; rr < 4; rr++) {
        int row = wid * 4 + rr;
        float s = 0.f, s2 = 0.f;
#pragma unroll
        for (int j = 0; j < 8; j++) {
            float x = tile[row][lane + 32 * j];
            s += x; s2 += x * x;
        }
#pragma unroll
        for (int off = 16; off; off >>= 1) {
            s  += __shfl_xor(s,  off, 32);
            s2 += __shfl_xor(s2, off, 32);
        }
        float mean = s * (1.f / 256.f);
        float var  = s2 * (1.f / 256.f) - mean * mean;
        float rstd = rsqrtf(var + 1e-5f);
#pragma unroll
        for (int j = 0; j < 8; j++) {
            int c = lane + 32 * j;
            tile[row][c] = (tile[row][c] - mean) * rstd;
        }
    }
    __syncthreads();
    int b = t0 >> 12, nb = t0 & (N_ - 1);
#pragma unroll 4
    for (int i = 0; i < 32; i++) {
        int idx = threadIdx.x + 256 * i;
        int c = idx >> 5, nl = idx & 31;
        outp[((size_t)b * C_ + c) * N_ + nb + nl] = tile[nl][c] * g[c] + bb[c];
    }
}

extern "C" void kernel_launch(void* const* d_in, const int* in_sizes, int n_in,
                              void* d_out, int out_size, void* d_ws, size_t ws_size,
                              hipStream_t stream) {
    (void)in_sizes; (void)n_in; (void)out_size; (void)ws_size;
    const float* feat  = (const float*)d_in[0];
    const float* dp    = (const float*)d_in[1];
    const float* Kmat  = (const float*)d_in[2];
    const float* pw    = (const float*)d_in[3];
    const float* pb    = (const float*)d_in[4];
    const float* fc1w  = (const float*)d_in[5];
    const float* fc1b  = (const float*)d_in[6];
    const float* fc2w  = (const float*)d_in[7];
    const float* fc2b  = (const float*)d_in[8];
    const float* projw = (const float*)d_in[9];
    const float* projb = (const float*)d_in[10];
    const float* lng   = (const float*)d_in[11];
    const float* lnb   = (const float*)d_in[12];
    const int*   strd  = (const int*)d_in[13];
    float* outp = (float*)d_out;

    char* ws = (char*)d_ws;
    auto al = [](size_t x) { return (x + 255) & ~(size_t)255; };
    size_t oXf = 0;
    size_t oH1 = al(oXf + (size_t)M_ * FUS_ * 2);
    size_t oH2 = al(oH1 + (size_t)M_ * FF_ * 2);
    size_t oY  = al(oH2 + (size_t)M_ * FUS_ * 2);
    size_t oW1 = al(oY  + (size_t)M_ * C_ * 4);
    size_t oW2 = al(oW1 + (size_t)FF_ * FUS_ * 2);
    size_t oWp = al(oW2 + (size_t)FUS_ * FF_ * 2);

    unsigned short* Xf = (unsigned short*)(ws + oXf);
    unsigned short* H1 = (unsigned short*)(ws + oH1);
    unsigned short* H2 = (unsigned short*)(ws + oH2);
    float*          Yp = (float*)(ws + oY);
    unsigned short* W1 = (unsigned short*)(ws + oW1);
    unsigned short* W2 = (unsigned short*)(ws + oW2);
    unsigned short* Wp = (unsigned short*)(ws + oWp);

    k_f32_to_bf16<<<(FF_ * FUS_ + 255) / 256, 256, 0, stream>>>(fc1w, W1, FF_ * FUS_);
    k_f32_to_bf16<<<(FUS_ * FF_ + 255) / 256, 256, 0, stream>>>(fc2w, W2, FUS_ * FF_);
    k_f32_to_bf16<<<(C_ * FUS_ + 255) / 256, 256, 0, stream>>>(projw, Wp, C_ * FUS_);

    k_feat_transpose<<<dim3(N_ / 32, C_ / 32, B_), dim3(32, 8), 0, stream>>>(feat, Xf);
    k_depth_pts<<<M_ / 8, 256, 0, stream>>>(dp, Kmat, pw, pb, strd, Xf);

    // fc1: [M,288] x [1024,288]^T -> relu -> H1 bf16
    k_gemm_bf16<<<dim3(FF_ / 128, M_ / 128), 256, 0, stream>>>(
        Xf, W1, fc1b, H1, nullptr, nullptr, FF_, FUS_, 0);
    // fc2: [M,1024] x [288,1024]^T -> H2 bf16
    k_gemm_bf16<<<dim3((FUS_ + 127) / 128, M_ / 128), 256, 0, stream>>>(
        H1, W2, fc2b, H2, nullptr, nullptr, FUS_, FF_, 1);
    // proj: [M,288] x [256,288]^T + resid -> Yp f32
    k_gemm_bf16<<<dim3(C_ / 128, M_ / 128), 256, 0, stream>>>(
        H2, Wp, projb, nullptr, Yp, Xf, C_, FUS_, 2);

    k_ln_transpose<<<M_ / 32, 256, 0, stream>>>(Yp, lng, lnb, outp);
}
